// NPairAngularLoss_26147760898822
// MI455X (gfx1250) — matrix operand order repaired and checked by reading, not compiled, not run on hardware
//
#include <hip/hip_runtime.h>
#include <hip/hip_bf16.h>
#include <math.h>

// ---- types matching the gfx1250 WMMA builtin signatures ----
typedef __bf16 bf16;
typedef bf16  v16bf __attribute__((ext_vector_type(16)));
typedef bf16  v8bf  __attribute__((ext_vector_type(8)));
typedef bf16  v4bf  __attribute__((ext_vector_type(4)));
typedef float v8f   __attribute__((ext_vector_type(8)));

#define GAMMA        0.001f
#define FACTOR_CONST 1.0f
#define W_COEF       1.0f

// ------------------------------------------------------------------
// 1) f32 -> bf16 conversion of X (vectorized 4-wide)
// ------------------------------------------------------------------
__global__ __launch_bounds__(256) void k_cvt(const float* __restrict__ x,
                                             bf16* __restrict__ xh, int n4) {
  int i = blockIdx.x * blockDim.x + threadIdx.x;
  if (i >= n4) return;
  float4 f = reinterpret_cast<const float4*>(x)[i];
  v4bf o;
  o.x = (bf16)f.x; o.y = (bf16)f.y; o.z = (bf16)f.z; o.w = (bf16)f.w;
  reinterpret_cast<v4bf*>(xh)[i] = o;
}

// ------------------------------------------------------------------
// 2) pos[q] = first k with targets[k]==targets[q] && k!=q  (0 if none)
// ------------------------------------------------------------------
__global__ __launch_bounds__(256) void k_pos(const int* __restrict__ tgt,
                                             int* __restrict__ pos, int Bn) {
  int q = blockIdx.x * blockDim.x + threadIdx.x;
  if (q >= Bn) return;
  int t = tgt[q];
  int p = 0;
  for (int k = 0; k < Bn; ++k) {
    if (k != q && tgt[k] == t) { p = k; break; }
  }
  pos[q] = p;
}

// ------------------------------------------------------------------
// 3) W = GAMMA * Xh * Xh^T  via v_wmma_f32_16x16x32_bf16
//    Each wave computes a 32(M) x 64(N) tile: 2 A frags x 4 B frags
//    -> 8 WMMA per 32-wide K step, ~35 FLOP/B from L2.
//
//    A (16x32 bf16) lane layout as loaded here (contiguous k, same as B):
//      lane L: M = L&15; elems 0..7  -> K = 16*(L>=16) + i
//              elems 8..15 -> K = 16*(L>=16) + i
//      => two contiguous 8-elem (16B) chunks at K = 16*hi and 16*hi+8.
//    B (32x16 bf16) lane layout as loaded here (not the ISA table order):
//      lane L: N = L&15; elem i -> K = i + 16*(L>=16)
//      => one contiguous 16-elem (32B) chunk at K = 16*hi.
//    C/D (16x16 f32): VGPR r, lanes 0-15 -> (M=r,   N=lane)
//                             lanes 16-31 -> (M=r+8, N=lane-16)
// ------------------------------------------------------------------
__global__ __launch_bounds__(256) void k_gemm(const bf16* __restrict__ Xh,
                                              float* __restrict__ W,
                                              int Bn, int D) {
  const int lane = threadIdx.x & 31;
  const int wave = threadIdx.x >> 5;
  const int groupsN = Bn >> 6;               // 64 columns per wave job
  const int jobs = (Bn >> 5) * groupsN;      // 32 rows per wave job
  int job = blockIdx.x * 8 + wave;
  if (job >= jobs) return;                   // wave-uniform: EXEC stays all-1s

  const int tm = job / groupsN;
  const int tn = job % groupsN;
  const int m0 = tm << 5;
  const int n0 = tn << 6;
  const int hi = lane >> 4;                  // 0 or 1
  const int lr = lane & 15;

  const bf16* __restrict__ Arow0 = Xh + (size_t)(m0 + lr) * D;
  const bf16* __restrict__ Arow1 = Arow0 + (size_t)16 * D;
  const bf16* __restrict__ Brow  = Xh + (size_t)(n0 + lr) * D;
  const size_t tstride = (size_t)16 * D;     // 16 rows ahead per B tile

  v8f acc[2][4];
#pragma unroll
  for (int mt = 0; mt < 2; ++mt)
#pragma unroll
    for (int t = 0; t < 4; ++t) acc[mt][t] = (v8f){};

  for (int k0 = 0; k0 < D; k0 += 32) {
    // near-cache prefetch a few K-steps ahead (emits global_prefetch_b8)
    __builtin_prefetch(Arow0 + k0 + 128, 0, 3);
    __builtin_prefetch(Brow + k0 + 128, 0, 3);

    // A fragments for the two 16-row M tiles: two 16B chunks, concatenated
    v8bf a0lo = *reinterpret_cast<const v8bf*>(Arow0 + k0 + 16 * hi);
    v8bf a0hi = *reinterpret_cast<const v8bf*>(Arow0 + k0 + 8 + 16 * hi);
    v8bf a1lo = *reinterpret_cast<const v8bf*>(Arow1 + k0 + 16 * hi);
    v8bf a1hi = *reinterpret_cast<const v8bf*>(Arow1 + k0 + 8 + 16 * hi);
    v16bf a0 = __builtin_shufflevector(a0lo, a0hi,
                                       0, 1, 2, 3, 4, 5, 6, 7,
                                       8, 9, 10, 11, 12, 13, 14, 15);
    v16bf a1 = __builtin_shufflevector(a1lo, a1hi,
                                       0, 1, 2, 3, 4, 5, 6, 7,
                                       8, 9, 10, 11, 12, 13, 14, 15);

    // B fragments: one contiguous 32B chunk per N tile
    const bf16* bp = Brow + k0 + 16 * hi;
#pragma unroll
    for (int t = 0; t < 4; ++t) {
      v16bf b = *reinterpret_cast<const v16bf*>(bp + (size_t)t * tstride);
      acc[0][t] = __builtin_amdgcn_wmma_f32_16x16x32_bf16(
          false, a0, false, b, (short)0, acc[0][t], false, false);
      acc[1][t] = __builtin_amdgcn_wmma_f32_16x16x32_bf16(
          false, a1, false, b, (short)0, acc[1][t], false, false);
    }
  }

  // scale by GAMMA and scatter to W per the C/D layout
#pragma unroll
  for (int mt = 0; mt < 2; ++mt) {
#pragma unroll
    for (int t = 0; t < 4; ++t) {
#pragma unroll
      for (int r = 0; r < 8; ++r) {
        W[(size_t)(m0 + mt * 16 + r + 8 * hi) * Bn + (n0 + t * 16 + lr)] =
            GAMMA * acc[mt][t][r];
      }
    }
  }
}

// ------------------------------------------------------------------
// 4) per-query masked reductions:
//    l1[q] = log1p( sum_{k!=q,pos} exp(W[q,k] - W[q,pos]) )
//    l2[q] = log1p( sum_{k!=q,pos} exp((W[q,k]+W[pos,k])*fc
//                                       - W[q,pos]*(fc/2+0.5)) )
// ------------------------------------------------------------------
__global__ __launch_bounds__(256) void k_reduce(const float* __restrict__ W,
                                                const int* __restrict__ pos,
                                                float* __restrict__ l1,
                                                float* __restrict__ l2,
                                                int Bn) {
  const int q = blockIdx.x;
  const int p = pos[q];
  const float* __restrict__ wq = W + (size_t)q * Bn;
  const float* __restrict__ wp = W + (size_t)p * Bn;
  const float wpos = wq[p];
  const float c2 = wpos * (FACTOR_CONST * 0.5f + 0.5f);

  float s1 = 0.0f, s2 = 0.0f;
  for (int k = threadIdx.x; k < Bn; k += blockDim.x) {
    if (k == q || k == p) continue;
    float w = wq[k];
    s1 += __expf(w - wpos);
    s2 += __expf((w + wp[k]) * FACTOR_CONST - c2);
  }

  // wave32 reduce, then cross-wave via LDS
  for (int off = 16; off > 0; off >>= 1) {
    s1 += __shfl_down(s1, off);
    s2 += __shfl_down(s2, off);
  }
  __shared__ float sm1[8], sm2[8];
  if ((threadIdx.x & 31) == 0) {
    sm1[threadIdx.x >> 5] = s1;
    sm2[threadIdx.x >> 5] = s2;
  }
  __syncthreads();
  if (threadIdx.x == 0) {
    float t1 = 0.0f, t2 = 0.0f;
#pragma unroll
    for (int i = 0; i < 8; ++i) { t1 += sm1[i]; t2 += sm2[i]; }
    l1[q] = log1pf(t1);
    l2[q] = log1pf(t2);
  }
}

// ------------------------------------------------------------------
// 5) deterministic final mean: out = mean(l1) + W_COEF * mean(l2)
// ------------------------------------------------------------------
__global__ __launch_bounds__(256) void k_fin(const float* __restrict__ l1,
                                             const float* __restrict__ l2,
                                             float* __restrict__ out, int Bn) {
  __shared__ float s1[256], s2[256];
  float a = 0.0f, b = 0.0f;
  for (int q = threadIdx.x; q < Bn; q += 256) { a += l1[q]; b += l2[q]; }
  s1[threadIdx.x] = a; s2[threadIdx.x] = b;
  __syncthreads();
  for (int off = 128; off > 0; off >>= 1) {
    if (threadIdx.x < off) {
      s1[threadIdx.x] += s1[threadIdx.x + off];
      s2[threadIdx.x] += s2[threadIdx.x + off];
    }
    __syncthreads();
  }
  if (threadIdx.x == 0) {
    float inv = 1.0f / (float)Bn;
    out[0] = s1[0] * inv + W_COEF * (s2[0] * inv);
  }
}

// ------------------------------------------------------------------
extern "C" void kernel_launch(void* const* d_in, const int* in_sizes, int n_in,
                              void* d_out, int out_size, void* d_ws, size_t ws_size,
                              hipStream_t stream) {
  const float* x   = (const float*)d_in[0];
  const int*   tgt = (const int*)d_in[1];
  const int Bn = in_sizes[1];           // 4096
  const int D  = in_sizes[0] / Bn;      // 2048

  // workspace layout (all 256B-aligned)
  char* ws = (char*)d_ws;
  size_t off = 0;
  bf16* Xh = (bf16*)(ws + off);
  off += ((size_t)Bn * D * sizeof(bf16) + 255) & ~(size_t)255;
  float* W = (float*)(ws + off);
  off += ((size_t)Bn * Bn * sizeof(float) + 255) & ~(size_t)255;
  int* pos = (int*)(ws + off);
  off += ((size_t)Bn * sizeof(int) + 255) & ~(size_t)255;
  float* l1 = (float*)(ws + off);
  off += ((size_t)Bn * sizeof(float) + 255) & ~(size_t)255;
  float* l2 = (float*)(ws + off);

  // 1) convert X to bf16
  int n4 = (Bn * D) / 4;
  k_cvt<<<(n4 + 255) / 256, 256, 0, stream>>>(x, Xh, n4);

  // 2) positives
  k_pos<<<(Bn + 255) / 256, 256, 0, stream>>>(tgt, pos, Bn);

  // 3) WMMA GEMM: (Bn/32)*(Bn/64) wave jobs, 8 waves per block
  int jobs = (Bn >> 5) * (Bn >> 6);
  k_gemm<<<(jobs + 7) / 8, 256, 0, stream>>>(Xh, W, Bn, D);

  // 4) per-query reductions
  k_reduce<<<Bn, 256, 0, stream>>>(W, pos, l1, l2, Bn);

  // 5) final scalar
  k_fin<<<1, 256, 0, stream>>>(l1, l2, (float*)d_out, Bn);
}
